// STI_326417514720
// MI455X (gfx1250) — matrix operand order repaired and checked by reading, not compiled, not run on hardware
//
#include <hip/hip_runtime.h>
#include <hip/hip_bf16.h>

// ---------------------------------------------------------------------------
// CDNA5 (gfx1250) implementation of the dilated-conv + linear-attention stack.
// All matrix contractions run through v_wmma_f32_16x16x32_bf16 (fp32 accum),
// with double-buffered LDS tiles and division-free conv-tap staging.
// ---------------------------------------------------------------------------

typedef __bf16 bf16;
typedef __attribute__((ext_vector_type(16))) __bf16 v16bf;
typedef __attribute__((ext_vector_type(8)))  float  v8f;

#define B_    4
#define T_    8192
#define NODE_ 64
#define F_    256
#define L_    10
#define OUT_  48
#define H_    4
#define S_    4
#define EPS_  1e-5f
#define ALPHA_ 1.0f

#define BM 128
#define BN 128
#define BK 32
#define BKP 40   // padded LDS K-stride (bf16 elems) to spread banks

// ---------------------------------------------------------------------------
// Generic token-major WMMA GEMM:
//   C[b][m][n] = epilogue( sum_k A[b][m(+shift)][k] * W[k][n] )
// A is bf16 token-major (row stride lda). W is bf16 (Ktot x N).
// Supports multi-tap dilated conv via Kper/dil (row shift per K-slab).
// ---------------------------------------------------------------------------
struct GemmP {
  const bf16* A; const bf16* W; const float* bias;
  const float* Res; const float* mask; const float* rowScale;
  float* Cf; bf16* Cb;
  int M, N, Ktot, Kper, lda, Tlen, Z;
  int zDivA; long zA0, zA1; long zW;
  int zDivC; long zC0, zC1;
  long out_rs, out_cs;
  long zRes; long zScale;
  int dil; int act; float alpha; int maskFirst, maskLast;
};

__device__ __forceinline__ float apply_act(float v, int act) {
  if (act == 1) return v > 0.f ? v : 0.f;
  if (act == 2) return 1.f / (1.f + expf(-v));
  if (act == 3) return 0.5f * v * (1.f + erff(v * 0.70710678118654752f));
  return v;
}

__global__ __launch_bounds__(256) void k_gemm(GemmP p) {
  __shared__ __align__(16) bf16 As[2][BM][BKP];
  __shared__ __align__(16) bf16 Bs[2][BN][BKP];

  const int z = blockIdx.z;
  const int bidx = z / p.zDivA;
  const bf16* Ab = p.A + (long)bidx * p.zA0 + (long)(z % p.zDivA) * p.zA1;
  const bf16* Wb = p.W + (long)z * p.zW;
  const float* Resb = p.Res ? p.Res + (long)bidx * p.zRes : nullptr;
  const float* Mb   = p.mask ? p.mask + (long)bidx * p.Tlen : nullptr;
  const float* RSb  = p.rowScale ? p.rowScale + (long)z * p.zScale : nullptr;
  const long cOff = (long)(z / p.zDivC) * p.zC0 + (long)(z % p.zDivC) * p.zC1;
  float* Cf = p.Cf ? p.Cf + cOff : nullptr;
  bf16*  Cb = p.Cb ? p.Cb + cOff : nullptr;

  const int m0 = blockIdx.x * BM, n0 = blockIdx.y * BN;
  const int tid = threadIdx.x, wave = tid >> 5, lane = tid & 31;
  const int wm = (wave & 1) * 64, wn = (wave >> 1) * 32;
  const int hi = lane >> 4, lm = lane & 15;
  const int taps = p.Ktot / p.Kper;

  // ---- division-free A-staging state (per-thread) ----
  const int rA  = tid >> 1;
  const int khA = (tid & 1) * 16;
  int  kiA = khA;                                    // k index within tap slab
  long shiftA = -(long)(taps >> 1) * (long)p.dil;    // row shift of current tap
  const bool rowOkA = (m0 + rA) < p.M;

  // ---- B-staging state ----
  const int kbB = tid & 31, nbB = (tid >> 5) * 16;
  int kkB = kbB;

  uint4 a0, a1, b0, b1;

  auto fetchA = [&]() {
    a0 = uint4{0u,0u,0u,0u}; a1 = uint4{0u,0u,0u,0u};
    const long row = (long)(m0 + rA) + shiftA;
    if (rowOkA && row >= 0 && row < p.Tlen) {
      const uint4* s = (const uint4*)(Ab + row * (long)p.lda + kiA);
      a0 = s[0]; a1 = s[1];
      __builtin_prefetch((const void*)(s + 4), 0, 0);   // speculative next K-step
    }
    kiA += BK;
    if (kiA >= p.Kper) { kiA -= p.Kper; shiftA += p.dil; }
  };
  auto fetchB = [&]() {
    const uint4* s = (const uint4*)(Wb + (long)kkB * p.N + (n0 + nbB));
    b0 = s[0]; b1 = s[1];
    kkB += BK;
  };
  auto stageA = [&](int buf) {
    uint4* d = (uint4*)&As[buf][rA][khA];
    d[0] = a0; d[1] = a1;
  };
  auto stageB = [&](int buf) {
    bf16 tmp[16];
    *(uint4*)&tmp[0] = b0; *(uint4*)&tmp[8] = b1;
#pragma unroll
    for (int j = 0; j < 16; ++j) Bs[buf][nbB + j][kbB] = tmp[j];
  };

  v8f acc[4][2];
#pragma unroll
  for (int i = 0; i < 4; ++i)
#pragma unroll
    for (int j = 0; j < 2; ++j)
#pragma unroll
      for (int r = 0; r < 8; ++r) acc[i][j][r] = 0.f;

  const int nIter = p.Ktot / BK;
  fetchA(); fetchB();
  stageA(0); stageB(0);
  __syncthreads();

  for (int it = 0; it < nIter; ++it) {
    const int cur = it & 1;
    const bool more = (it + 1) < nIter;
    if (more) { fetchA(); fetchB(); }

    // ---- fragments per CDNA5 16-bit A(16x32) / B(32x16) layouts
    v16bf af[4], bfr[2];
#pragma unroll
    for (int i = 0; i < 4; ++i) {
      const bf16* ap = &As[cur][wm + i * 16 + lm][0];
#pragma unroll
      for (int e = 0; e < 8; ++e) {
        af[i][e]     = ap[hi * 8 + e];
        af[i][8 + e] = ap[16 + hi * 8 + e];
      }
    }
#pragma unroll
    for (int j = 0; j < 2; ++j) {
      const bf16* bp = &Bs[cur][wn + j * 16 + lm][0];
#pragma unroll
      for (int e = 0; e < 16; ++e) bfr[j][e] = bp[e < 8 ? hi * 8 + e : 16 + hi * 8 + (e - 8)];
    }
#pragma unroll
    for (int i = 0; i < 4; ++i)
#pragma unroll
      for (int j = 0; j < 2; ++j)
        acc[i][j] = __builtin_amdgcn_wmma_f32_16x16x32_bf16(
            false, af[i], false, bfr[j], (short)0, acc[i][j], false, false);

    if (more) { stageA(cur ^ 1); stageB(cur ^ 1); }
    __syncthreads();
  }

  // ---- epilogue (C 16x16 f32 layout: lane n=lm, m=hi*8+r)
#pragma unroll
  for (int i = 0; i < 4; ++i)
#pragma unroll
    for (int j = 0; j < 2; ++j)
#pragma unroll
      for (int r = 0; r < 8; ++r) {
        const int m = m0 + wm + i * 16 + hi * 8 + r;
        const int n = n0 + wn + j * 16 + lm;
        if (m < p.M && n < p.N) {
          float v = acc[i][j][r];
          if (p.bias) v += p.bias[n];
          v = apply_act(v, p.act);
          if (RSb) v *= RSb[m];
          if (p.maskFirst && Mb) v *= Mb[m];
          v *= p.alpha;
          if (Resb) v += Resb[(long)m * p.N + n];
          if (p.maskLast && Mb) v *= Mb[m];
          const long idx = (long)m * p.out_rs + (long)n * p.out_cs;
          if (Cf) Cf[idx] = v;
          if (Cb) Cb[idx] = (bf16)v;
        }
      }
}

// ---------------------------------------------------------------------------
// A^T * B over long T (KV accumulation, SFI score matrix).
//   C[z][m][n] += sum_t A[z][t][m] * B[z][t][n]     (atomic fp32 accumulate)
// ---------------------------------------------------------------------------
struct AtP {
  const bf16* Am; const bf16* Bm; float* C;
  int M, N, lda, ldb, T, chunks;
  int zDiv; long zA0, zA1, zB0, zB1, zC;
};

__global__ __launch_bounds__(256) void k_atgemm(AtP p) {
  __shared__ __align__(16) bf16 As[2][BM][BKP];
  __shared__ __align__(16) bf16 Bs[2][BN][BKP];

  const int zz = blockIdx.z / p.chunks, ch = blockIdx.z % p.chunks;
  const bf16* Ab = p.Am + (long)(zz / p.zDiv) * p.zA0 + (long)(zz % p.zDiv) * p.zA1;
  const bf16* Bb = p.Bm + (long)(zz / p.zDiv) * p.zB0 + (long)(zz % p.zDiv) * p.zB1;
  float* Cz = p.C + (long)zz * p.zC;
  const int clen = p.T / p.chunks;
  const int t0 = ch * clen;

  const int m0 = blockIdx.x * BM, n0 = blockIdx.y * BN;
  const int tid = threadIdx.x, wave = tid >> 5, lane = tid & 31;
  const int wm = (wave & 1) * 64, wn = (wave >> 1) * 32;
  const int hi = lane >> 4, lm = lane & 15;

  const int kb = tid & 31, cb = (tid >> 5) * 16;
  long rowF = t0 + kb;
  uint4 a0, a1, b0, b1;

  auto fetchAB = [&]() {
    const uint4* sa = (const uint4*)(Ab + rowF * (long)p.lda + (m0 + cb));
    a0 = sa[0]; a1 = sa[1];
    const uint4* sb = (const uint4*)(Bb + rowF * (long)p.ldb + (n0 + cb));
    b0 = sb[0]; b1 = sb[1];
    __builtin_prefetch((const void*)(sa + BK * p.lda / 8), 0, 0);
    rowF += BK;
  };
  auto stageAB = [&](int buf) {
    bf16 ta[16], tb[16];
    *(uint4*)&ta[0] = a0; *(uint4*)&ta[8] = a1;
    *(uint4*)&tb[0] = b0; *(uint4*)&tb[8] = b1;
#pragma unroll
    for (int j = 0; j < 16; ++j) {
      As[buf][cb + j][kb] = ta[j];
      Bs[buf][cb + j][kb] = tb[j];
    }
  };

  v8f acc[4][2];
#pragma unroll
  for (int i = 0; i < 4; ++i)
#pragma unroll
    for (int j = 0; j < 2; ++j)
#pragma unroll
      for (int r = 0; r < 8; ++r) acc[i][j][r] = 0.f;

  const int nIter = clen / BK;
  fetchAB();
  stageAB(0);
  __syncthreads();

  for (int it = 0; it < nIter; ++it) {
    const int cur = it & 1;
    const bool more = (it + 1) < nIter;
    if (more) fetchAB();

    v16bf af[4], bfr[2];
#pragma unroll
    for (int i = 0; i < 4; ++i) {
      const bf16* ap = &As[cur][wm + i * 16 + lm][0];
#pragma unroll
      for (int e = 0; e < 8; ++e) {
        af[i][e]     = ap[hi * 8 + e];
        af[i][8 + e] = ap[16 + hi * 8 + e];
      }
    }
#pragma unroll
    for (int j = 0; j < 2; ++j) {
      const bf16* bp = &Bs[cur][wn + j * 16 + lm][0];
#pragma unroll
      for (int e = 0; e < 16; ++e) bfr[j][e] = bp[e < 8 ? hi * 8 + e : 16 + hi * 8 + (e - 8)];
    }
#pragma unroll
    for (int i = 0; i < 4; ++i)
#pragma unroll
      for (int j = 0; j < 2; ++j)
        acc[i][j] = __builtin_amdgcn_wmma_f32_16x16x32_bf16(
            false, af[i], false, bfr[j], (short)0, acc[i][j], false, false);

    if (more) stageAB(cur ^ 1);
    __syncthreads();
  }

#pragma unroll
  for (int i = 0; i < 4; ++i)
#pragma unroll
    for (int j = 0; j < 2; ++j)
#pragma unroll
      for (int r = 0; r < 8; ++r) {
        const int m = m0 + wm + i * 16 + hi * 8 + r;
        const int n = n0 + wn + j * 16 + lm;
        if (m < p.M && n < p.N)
          atomicAdd(&Cz[(long)m * p.N + n], acc[i][j][r]);
      }
}

// ---------------------------------------------------------------------------
// Small helper kernels
// ---------------------------------------------------------------------------
__global__ void k_memset_f(float* p, long n) {
  for (long i = blockIdx.x * 256L + threadIdx.x; i < n; i += gridDim.x * 256L) p[i] = 0.f;
}
__global__ void k_cast_bf(const float* s, bf16* d, long n) {
  for (long i = blockIdx.x * 256L + threadIdx.x; i < n; i += gridDim.x * 256L) d[i] = (bf16)s[i];
}
// src (N,K) fp32 -> dst (K,N) bf16
__global__ void k_castT(const float* src, bf16* dst, int N, int K) {
  long i = blockIdx.x * 256L + threadIdx.x;
  if (i < (long)N * K) {
    int k = (int)(i / N), n = (int)(i % N);
    dst[i] = (bf16)src[(long)n * K + k];
  }
}
// ff_w layer slice (F,F,3) -> packed (3F, F): dst[(tap*F+i)*F + o] = w[(o*F+i)*3 + tap]
__global__ void k_pack_conv(const float* w, bf16* dst) {
  long i = blockIdx.x * 256L + threadIdx.x;
  if (i < 768L * F_) {
    int o = (int)(i % F_); int ki = (int)(i / F_);
    int tap = ki / F_, ii = ki % F_;
    dst[i] = (bf16)w[((long)o * F_ + ii) * 3 + tap];
  }
}
// conv_in: 3->5 channel mix; writes S bf16 slices (token-major) + ft bf16
__global__ void k_conv_in(const float* x, const float* w, const float* b,
                          bf16* fs, bf16* ft) {
  long i = blockIdx.x * 256L + threadIdx.x;
  if (i >= (long)B_ * T_ * NODE_) return;
  int n = (int)(i % NODE_); long bt = i / NODE_;
  int bb = (int)(bt / T_); int t = (int)(bt % T_);
  float xi0 = x[(((long)bb * 3 + 0) * T_ + t) * NODE_ + n];
  float xi1 = x[(((long)bb * 3 + 1) * T_ + t) * NODE_ + n];
  float xi2 = x[(((long)bb * 3 + 2) * T_ + t) * NODE_ + n];
#pragma unroll
  for (int o = 0; o < S_ + 1; ++o) {
    float v = b[o] + w[o * 3 + 0] * xi0 + w[o * 3 + 1] * xi1 + w[o * 3 + 2] * xi2;
    if (o < S_) fs[(((long)o * B_ + bb) * T_ + t) * NODE_ + n] = (bf16)v;
    else        ft[i] = (bf16)v;
  }
}
// instance norm over T per (b,f); output bf16 token-major
__global__ void k_inorm(const float* x, bf16* qn) {
  int id = blockIdx.x; int bb = id / F_, f = id % F_;
  const float* xp = x + (long)bb * T_ * F_ + f;
  __shared__ float s1[256], s2[256];
  float a = 0.f, q = 0.f;
  for (int t = threadIdx.x; t < T_; t += 256) {
    float v = xp[(long)t * F_]; a += v; q += v * v;
  }
  s1[threadIdx.x] = a; s2[threadIdx.x] = q; __syncthreads();
  for (int o = 128; o > 0; o >>= 1) {
    if (threadIdx.x < o) { s1[threadIdx.x] += s1[threadIdx.x + o]; s2[threadIdx.x] += s2[threadIdx.x + o]; }
    __syncthreads();
  }
  float m = s1[0] * (1.f / T_);
  float var = s2[0] * (1.f / T_) - m * m;
  float r = rsqrtf(var + EPS_);
  bf16* qp = qn + (long)bb * T_ * F_ + f;
  for (int t = threadIdx.x; t < T_; t += 256)
    qp[(long)t * F_] = (bf16)((xp[(long)t * F_] - m) * r);
}
// Ksum[b*F+f] = sum_t K[b,t,f]
__global__ void k_ksum(const bf16* Ks, float* out) {
  int id = blockIdx.x; int bb = id / F_, f = id % F_;
  const bf16* p = Ks + (long)bb * T_ * F_ + f;
  __shared__ float red[256];
  float a = 0.f;
  for (int t = threadIdx.x; t < T_; t += 256) a += (float)p[(long)t * F_];
  red[threadIdx.x] = a; __syncthreads();
  for (int o = 128; o > 0; o >>= 1) {
    if (threadIdx.x < o) red[threadIdx.x] += red[threadIdx.x + o];
    __syncthreads();
  }
  if (threadIdx.x == 0) out[id] = red[0];
}
// Z[b,h,t] = 1 / sum_d Q[b,t,h*64+d]*(Ksum[b,h*64+d]+1e-6)
__global__ void k_zinv(const bf16* Q, const float* ks, float* Z) {
  long i = blockIdx.x * 256L + threadIdx.x;
  if (i >= (long)B_ * H_ * T_) return;
  int t = (int)(i % T_); long bh = i / T_;
  int bb = (int)(bh / H_); int h = (int)(bh % H_);
  const bf16* qp = Q + ((long)bb * T_ + t) * F_ + h * 64;
  const float* kp = ks + bb * F_ + h * 64;
  float s = 0.f;
#pragma unroll 8
  for (int d = 0; d < 64; ++d) s += (float)qp[d] * (kp[d] + 1e-6f);
  Z[i] = 1.f / s;
}
// softmax over d of sc[b,c,:] * (1/T); write transposed bf16 attnT[b,d,c]
__global__ void k_softmax(const float* sc, bf16* attnT) {
  int bc = blockIdx.x; int bb = bc / F_, c = bc % F_;
  int d = threadIdx.x;
  float v = sc[((long)bb * F_ + c) * F_ + d] * (1.f / T_);
  __shared__ float red[256];
  red[d] = v; __syncthreads();
  for (int o = 128; o > 0; o >>= 1) {
    if (d < o) red[d] = fmaxf(red[d], red[d + o]);
    __syncthreads();
  }
  float mx = red[0]; __syncthreads();
  float e = expf(v - mx);
  red[d] = e; __syncthreads();
  for (int o = 128; o > 0; o >>= 1) {
    if (d < o) red[d] += red[d + o];
    __syncthreads();
  }
  float sm = red[0];
  attnT[((long)bb * F_ + d) * F_ + c] = (bf16)(e / sm);
}

// ---------------------------------------------------------------------------
// Host orchestration
// ---------------------------------------------------------------------------
static inline void launch_gemm(hipStream_t st, const GemmP& p) {
  dim3 g((p.M + BM - 1) / BM, (p.N + BN - 1) / BN, p.Z);
  k_gemm<<<g, 256, 0, st>>>(p);
}
static inline void launch_at(hipStream_t st, const AtP& p, int Z) {
  dim3 g((p.M + BM - 1) / BM, (p.N + BN - 1) / BN, Z * p.chunks);
  k_atgemm<<<g, 256, 0, st>>>(p);
}

extern "C" void kernel_launch(void* const* d_in, const int* in_sizes, int n_in,
                              void* d_out, int out_size, void* d_ws, size_t ws_size,
                              hipStream_t stream) {
  const float* x         = (const float*)d_in[0];
  const float* mask      = (const float*)d_in[1];
  const float* conv_in_w = (const float*)d_in[2];
  const float* conv_in_b = (const float*)d_in[3];
  const float* conv_t_w  = (const float*)d_in[4];
  const float* conv_t_b  = (const float*)d_in[5];
  const float* sfi_cs_w  = (const float*)d_in[6];
  const float* sfi_cs_b  = (const float*)d_in[7];
  const float* sfi_ff1_w = (const float*)d_in[8];
  const float* sfi_ff1_b = (const float*)d_in[9];
  const float* sfi_ff2_w = (const float*)d_in[10];
  const float* sfi_ff2_b = (const float*)d_in[11];
  const float* ff_w      = (const float*)d_in[12];
  const float* ff_b      = (const float*)d_in[13];
  const float* qw = (const float*)d_in[14]; const float* qb = (const float*)d_in[15];
  const float* kw = (const float*)d_in[16]; const float* kb = (const float*)d_in[17];
  const float* vw = (const float*)d_in[18]; const float* vb = (const float*)d_in[19];
  const float* ow = (const float*)d_in[20]; const float* ob = (const float*)d_in[21];
  const float* co_w = (const float*)d_in[22]; const float* co_b = (const float*)d_in[23];
  const float* out_w = (const float*)d_in[24]; const float* out_b = (const float*)d_in[25];

  char* ws = (char*)d_ws;
  size_t off = 0;
  auto ALLOC = [&](size_t bytes) -> void* {
    void* p = ws + off;
    off += (bytes + 255) & ~(size_t)255;
    return p;
  };
  const size_t BTF = (size_t)B_ * T_ * F_;
  float* fst_f  = (float*)ALLOC(BTF * 4);
  bf16*  fst_b  = (bf16*) ALLOC(BTF * 2);
  float* conv_f = (float*)ALLOC(BTF * 4);
  float* att_f  = (float*)ALLOC(BTF * 4);
  bf16*  att_b  = (bf16*) ALLOC(BTF * 2);
  bf16*  s1 = (bf16*)ALLOC(BTF * 2);  // qn  / fsc
  bf16*  s2 = (bf16*)ALLOC(BTF * 2);  // Q   / cross
  bf16*  s3 = (bf16*)ALLOC(BTF * 2);  // K   / h1
  bf16*  s4 = (bf16*)ALLOC(BTF * 2);  // V
  bf16*  s5 = (bf16*)ALLOC(BTF * 2);  // o (heads assembled)
  bf16*  fs_b = (bf16*)ALLOC((size_t)S_ * B_ * T_ * NODE_ * 2);
  bf16*  ft_b = (bf16*)ALLOC((size_t)B_ * T_ * NODE_ * 2);
  float* KV_f = (float*)ALLOC((size_t)B_ * H_ * 64 * 64 * 4);
  bf16*  KV_b = (bf16*) ALLOC((size_t)B_ * H_ * 64 * 64 * 2);
  float* Ksum = (float*)ALLOC((size_t)B_ * F_ * 4);
  float* Zb   = (float*)ALLOC((size_t)B_ * H_ * T_ * 4);
  float* sc_f = (float*)ALLOC((size_t)B_ * F_ * F_ * 4);
  bf16*  attnT= (bf16*) ALLOC((size_t)B_ * F_ * F_ * 2);
  bf16*  Wconv = (bf16*)ALLOC((size_t)L_ * 768 * F_ * 2);
  bf16*  Wq = (bf16*)ALLOC((size_t)L_ * F_ * F_ * 2);
  bf16*  Wk = (bf16*)ALLOC((size_t)L_ * F_ * F_ * 2);
  bf16*  Wv = (bf16*)ALLOC((size_t)L_ * F_ * F_ * 2);
  bf16*  Wo = (bf16*)ALLOC((size_t)L_ * F_ * F_ * 2);
  bf16*  Wco= (bf16*)ALLOC((size_t)L_ * F_ * F_ * 2);
  bf16*  Wff1 = (bf16*)ALLOC((size_t)S_ * F_ * F_ * 2);
  bf16*  Wff2 = (bf16*)ALLOC((size_t)S_ * F_ * F_ * 2);
  bf16*  WconvT = (bf16*)ALLOC((size_t)NODE_ * F_ * 2);
  bf16*  Wcs  = (bf16*)ALLOC((size_t)S_ * NODE_ * F_ * 2);
  bf16*  Wout = (bf16*)ALLOC((size_t)F_ * OUT_ * 2);
  (void)ALLOC(8192);   // tail pad: unguarded vector loads may overread tiles

  // ---- weight preprocessing (bf16 transposed layouts) ----
  {
    int g = (F_ * NODE_ + 255) / 256;
    k_castT<<<g, 256, 0, stream>>>(conv_t_w, WconvT, F_, NODE_);
    for (int s = 0; s < S_; ++s)
      k_castT<<<g, 256, 0, stream>>>(sfi_cs_w + (size_t)s * F_ * NODE_,
                                     Wcs + (size_t)s * NODE_ * F_, F_, NODE_);
    int gf = (F_ * F_ + 255) / 256;
    for (int s = 0; s < S_; ++s) {
      k_castT<<<gf, 256, 0, stream>>>(sfi_ff1_w + (size_t)s * F_ * F_,
                                      Wff1 + (size_t)s * F_ * F_, F_, F_);
      k_castT<<<gf, 256, 0, stream>>>(sfi_ff2_w + (size_t)s * F_ * F_,
                                      Wff2 + (size_t)s * F_ * F_, F_, F_);
    }
    int gc = (768 * F_ + 255) / 256;
    for (int l = 0; l < L_; ++l) {
      k_pack_conv<<<gc, 256, 0, stream>>>(ff_w + (size_t)l * F_ * F_ * 3,
                                          Wconv + (size_t)l * 768 * F_);
      k_castT<<<gf, 256, 0, stream>>>(qw + (size_t)l * F_ * F_, Wq + (size_t)l * F_ * F_, F_, F_);
      k_castT<<<gf, 256, 0, stream>>>(kw + (size_t)l * F_ * F_, Wk + (size_t)l * F_ * F_, F_, F_);
      k_castT<<<gf, 256, 0, stream>>>(vw + (size_t)l * F_ * F_, Wv + (size_t)l * F_ * F_, F_, F_);
      k_castT<<<gf, 256, 0, stream>>>(ow + (size_t)l * F_ * F_, Wo + (size_t)l * F_ * F_, F_, F_);
      k_castT<<<gf, 256, 0, stream>>>(co_w + (size_t)l * F_ * F_, Wco + (size_t)l * F_ * F_, F_, F_);
    }
    int go = (F_ * OUT_ + 255) / 256;
    k_castT<<<go, 256, 0, stream>>>(out_w, Wout, OUT_, F_);
  }

  // ---- input projection + conv_t GEMM -> fst ----
  k_conv_in<<<(B_ * T_ * NODE_ + 255) / 256, 256, 0, stream>>>(x, conv_in_w, conv_in_b, fs_b, ft_b);
  {
    GemmP p = {};
    p.A = ft_b; p.W = WconvT; p.bias = conv_t_b;
    p.Cf = fst_f; p.Cb = fst_b;
    p.M = T_; p.N = F_; p.Ktot = NODE_; p.Kper = NODE_; p.lda = NODE_; p.Tlen = T_;
    p.Z = B_; p.zDivA = 1; p.zA0 = (long)T_ * NODE_;
    p.zDivC = 1; p.zC0 = (long)T_ * F_;
    p.out_rs = F_; p.out_cs = 1; p.alpha = 1.f;
    launch_gemm(stream, p);
  }

  int cnt = 0;
  for (int l = 0; l < L_; ++l) {
    if (l == 1 || l == 3 || l == 5 || l == 7) {
      // ================= SFI block =================
      const int c = cnt++;
      { // fsc = fs @ csw^T + csb  -> s1 (bf16)
        GemmP p = {};
        p.A = fs_b + (size_t)c * B_ * T_ * NODE_; p.W = Wcs + (size_t)c * NODE_ * F_;
        p.bias = sfi_cs_b + c * F_;
        p.Cb = s1;
        p.M = T_; p.N = F_; p.Ktot = NODE_; p.Kper = NODE_; p.lda = NODE_; p.Tlen = T_;
        p.Z = B_; p.zDivA = 1; p.zA0 = (long)T_ * NODE_;
        p.zDivC = 1; p.zC0 = (long)T_ * F_;
        p.out_rs = F_; p.out_cs = 1; p.alpha = 1.f;
        launch_gemm(stream, p);
      }
      k_memset_f<<<256, 256, 0, stream>>>(sc_f, (long)B_ * F_ * F_);
      { // scores[b,c,d] = sum_t fsc[t,c]*fst[t,d]
        AtP p = {};
        p.Am = s1; p.Bm = fst_b; p.C = sc_f;
        p.M = F_; p.N = F_; p.lda = F_; p.ldb = F_; p.T = T_; p.chunks = 32;
        p.zDiv = 1; p.zA0 = (long)T_ * F_; p.zB0 = (long)T_ * F_; p.zC = (long)F_ * F_;
        launch_at(stream, p, B_);
      }
      k_softmax<<<B_ * F_, 256, 0, stream>>>(sc_f, attnT);
      { // cross = fst @ attn^T + fst -> s2 (bf16)
        GemmP p = {};
        p.A = fst_b; p.W = attnT; p.Res = fst_f;
        p.Cb = s2;
        p.M = T_; p.N = F_; p.Ktot = F_; p.Kper = F_; p.lda = F_; p.Tlen = T_;
        p.Z = B_; p.zDivA = 1; p.zA0 = (long)T_ * F_; p.zW = (long)F_ * F_;
        p.zDivC = 1; p.zC0 = (long)T_ * F_; p.zRes = (long)T_ * F_;
        p.out_rs = F_; p.out_cs = 1; p.alpha = 1.f;
        launch_gemm(stream, p);
      }
      { // h1 = gelu(cross @ f1w^T + f1b) -> s3
        GemmP p = {};
        p.A = s2; p.W = Wff1 + (size_t)c * F_ * F_; p.bias = sfi_ff1_b + c * F_;
        p.Cb = s3;
        p.M = T_; p.N = F_; p.Ktot = F_; p.Kper = F_; p.lda = F_; p.Tlen = T_;
        p.Z = B_; p.zDivA = 1; p.zA0 = (long)T_ * F_;
        p.zDivC = 1; p.zC0 = (long)T_ * F_;
        p.out_rs = F_; p.out_cs = 1; p.alpha = 1.f; p.act = 3;
        launch_gemm(stream, p);
      }
      { // fst = (h1 @ f2w^T + f2b + fst) * mask
        GemmP p = {};
        p.A = s3; p.W = Wff2 + (size_t)c * F_ * F_; p.bias = sfi_ff2_b + c * F_;
        p.Res = fst_f; p.mask = mask;
        p.Cf = fst_f; p.Cb = fst_b;
        p.M = T_; p.N = F_; p.Ktot = F_; p.Kper = F_; p.lda = F_; p.Tlen = T_;
        p.Z = B_; p.zDivA = 1; p.zA0 = (long)T_ * F_;
        p.zDivC = 1; p.zC0 = (long)T_ * F_; p.zRes = (long)T_ * F_;
        p.out_rs = F_; p.out_cs = 1; p.alpha = 1.f; p.maskLast = 1;
        launch_gemm(stream, p);
      }
    }
    // ================= attention module =================
    const int dil = 1 << l;
    { // conv_out = relu(dilated_conv(fst) + b) -> conv_f (fp32)
      GemmP p = {};
      p.A = fst_b; p.W = Wconv + (size_t)l * 768 * F_; p.bias = ff_b + l * F_;
      p.Cf = conv_f;
      p.M = T_; p.N = F_; p.Ktot = 768; p.Kper = F_; p.lda = F_; p.Tlen = T_;
      p.Z = B_; p.zDivA = 1; p.zA0 = (long)T_ * F_;
      p.zDivC = 1; p.zC0 = (long)T_ * F_;
      p.out_rs = F_; p.out_cs = 1; p.alpha = 1.f; p.act = 1; p.dil = dil;
      launch_gemm(stream, p);
    }
    k_inorm<<<B_ * F_, 256, 0, stream>>>(conv_f, s1);
    { // Q = sigmoid(qn @ qw^T + qb) -> s2
      GemmP p = {};
      p.A = s1; p.W = Wq + (size_t)l * F_ * F_; p.bias = qb + l * F_;
      p.Cb = s2;
      p.M = T_; p.N = F_; p.Ktot = F_; p.Kper = F_; p.lda = F_; p.Tlen = T_;
      p.Z = B_; p.zDivA = 1; p.zA0 = (long)T_ * F_;
      p.zDivC = 1; p.zC0 = (long)T_ * F_;
      p.out_rs = F_; p.out_cs = 1; p.alpha = 1.f; p.act = 2;
      launch_gemm(stream, p);
    }
    { // K = sigmoid(...) -> s3
      GemmP p = {};
      p.A = s1; p.W = Wk + (size_t)l * F_ * F_; p.bias = kb + l * F_;
      p.Cb = s3;
      p.M = T_; p.N = F_; p.Ktot = F_; p.Kper = F_; p.lda = F_; p.Tlen = T_;
      p.Z = B_; p.zDivA = 1; p.zA0 = (long)T_ * F_;
      p.zDivC = 1; p.zC0 = (long)T_ * F_;
      p.out_rs = F_; p.out_cs = 1; p.alpha = 1.f; p.act = 2;
      launch_gemm(stream, p);
    }
    { // V -> s4
      GemmP p = {};
      p.A = s1; p.W = Wv + (size_t)l * F_ * F_; p.bias = vb + l * F_;
      p.Cb = s4;
      p.M = T_; p.N = F_; p.Ktot = F_; p.Kper = F_; p.lda = F_; p.Tlen = T_;
      p.Z = B_; p.zDivA = 1; p.zA0 = (long)T_ * F_;
      p.zDivC = 1; p.zC0 = (long)T_ * F_;
      p.out_rs = F_; p.out_cs = 1; p.alpha = 1.f;
      launch_gemm(stream, p);
    }
    k_memset_f<<<64, 256, 0, stream>>>(KV_f, (long)B_ * H_ * 64 * 64);
    { // KV[b,h,d,e] = sum_t K[t,d]*V[t,e]
      AtP p = {};
      p.Am = s3; p.Bm = s4; p.C = KV_f;
      p.M = 64; p.N = 64; p.lda = F_; p.ldb = F_; p.T = T_; p.chunks = 32;
      p.zDiv = H_; p.zA0 = (long)T_ * F_; p.zA1 = 64; p.zB0 = (long)T_ * F_; p.zB1 = 64;
      p.zC = 64 * 64;
      launch_at(stream, p, B_ * H_);
    }
    k_cast_bf<<<64, 256, 0, stream>>>(KV_f, KV_b, (long)B_ * H_ * 64 * 64);
    k_ksum<<<B_ * F_, 256, 0, stream>>>(s3, Ksum);
    k_zinv<<<(B_ * H_ * T_ + 255) / 256, 256, 0, stream>>>(s2, Ksum, Zb);
    { // o[b,h,t,e] = (Q @ KV) * Z  -> s5 assembled (B,T,F)
      GemmP p = {};
      p.A = s2; p.W = KV_b; p.rowScale = Zb;
      p.Cb = s5;
      p.M = T_; p.N = 64; p.Ktot = 64; p.Kper = 64; p.lda = F_; p.Tlen = T_;
      p.Z = B_ * H_; p.zDivA = H_; p.zA0 = (long)T_ * F_; p.zA1 = 64;
      p.zW = 64 * 64;
      p.zDivC = H_; p.zC0 = (long)T_ * F_; p.zC1 = 64;
      p.out_rs = F_; p.out_cs = 1; p.zScale = T_; p.alpha = 1.f;
      launch_gemm(stream, p);
    }
    { // out = ALPHA * (o @ wo^T + ob) * mask + conv_out -> att_f/att_b
      GemmP p = {};
      p.A = s5; p.W = Wo + (size_t)l * F_ * F_; p.bias = ob + l * F_;
      p.Res = conv_f; p.mask = mask;
      p.Cf = att_f; p.Cb = att_b;
      p.M = T_; p.N = F_; p.Ktot = F_; p.Kper = F_; p.lda = F_; p.Tlen = T_;
      p.Z = B_; p.zDivA = 1; p.zA0 = (long)T_ * F_;
      p.zDivC = 1; p.zC0 = (long)T_ * F_; p.zRes = (long)T_ * F_;
      p.out_rs = F_; p.out_cs = 1; p.alpha = ALPHA_; p.maskFirst = 1;
      launch_gemm(stream, p);
    }
    { // fst = (fst + out @ co_w^T + co_b) * mask
      GemmP p = {};
      p.A = att_b; p.W = Wco + (size_t)l * F_ * F_; p.bias = co_b + l * F_;
      p.Res = fst_f; p.mask = mask;
      p.Cf = fst_f; p.Cb = fst_b;
      p.M = T_; p.N = F_; p.Ktot = F_; p.Kper = F_; p.lda = F_; p.Tlen = T_;
      p.Z = B_; p.zDivA = 1; p.zA0 = (long)T_ * F_;
      p.zDivC = 1; p.zC0 = (long)T_ * F_; p.zRes = (long)T_ * F_;
      p.out_rs = F_; p.out_cs = 1; p.alpha = 1.f; p.maskLast = 1;
      launch_gemm(stream, p);
    }
  }

  { // final: out[b,o,t] = (fst @ out_w^T + out_b) * mask, stored (B,OUT,T)
    GemmP p = {};
    p.A = fst_b; p.W = Wout; p.bias = out_b; p.mask = mask;
    p.Cf = (float*)d_out;
    p.M = T_; p.N = OUT_; p.Ktot = F_; p.Kper = F_; p.lda = F_; p.Tlen = T_;
    p.Z = B_; p.zDivA = 1; p.zA0 = (long)T_ * F_;
    p.zDivC = 1; p.zC0 = (long)OUT_ * T_;
    p.out_rs = 1; p.out_cs = T_; p.alpha = 1.f; p.maskLast = 1;
    launch_gemm(stream, p);
  }
}
